// XFeatSparseEncoder_69114613727362
// MI455X (gfx1250) — matrix operand order repaired and checked by reading, not compiled, not run on hardware
//
#include <hip/hip_runtime.h>
#include <math.h>

typedef __attribute__((ext_vector_type(16))) _Float16 v16h;
typedef __attribute__((ext_vector_type(8)))  float    v8f;
typedef unsigned int tdm_v4u __attribute__((ext_vector_type(4)));
typedef int          tdm_v8i __attribute__((ext_vector_type(8)));
typedef int          tdm_v4i __attribute__((ext_vector_type(4)));

#define B_       32
#define H8_      100
#define W8_      100
#define NPIX     (H8_ * W8_)        // 10000
#define C_LOG    65
#define HOUT     800
#define WOUT     800
#define NPIXOUT  (HOUT * WOUT)      // 640000
#define NMS_K_   32768
#define TOP_K_   4096
#define FEAT_C   64
#define TAU_     0.2f
#define CUBIC_A_ (-0.75f)

// Select element idx (0..7) of a WMMA f32 accumulator.
__device__ __forceinline__ float vsel8(v8f c, int idx) {
  float r;
  switch (idx & 7) {
    case 0: r = c[0]; break; case 1: r = c[1]; break;
    case 2: r = c[2]; break; case 3: r = c[3]; break;
    case 4: r = c[4]; break; case 5: r = c[5]; break;
    case 6: r = c[6]; break; default: r = c[7]; break;
  }
  return r;
}

// C/D 16x16 f32 layout: lanes 0-15 hold rows 0..7 (col=lane), lanes 16-31 hold
// rows 8..15 (col=lane-16). With every column of D equal (broadcast-column B),
// lane pair (m, m+16) recovers row m's value with one half-swap shuffle.
__device__ __forceinline__ float extract_row(v8f c, int lane, int m) {
  float own   = vsel8(c, m & 7);
  float other = __shfl_xor(own, 16, 32);
  return ((lane < 16) == (m < 8)) ? own : other;
}

// ---------------------------------------------------------------------------
// K1: softmax over 65 channels + 8x8 pixel-shuffle to kh (B,800,800).
// One wave handles 16 pixels; the 64-wide softmax denominator is computed with
// two V_WMMA_F32_16X16X32_F16 row-sum ops (B = ones), co-executing with the
// v_exp_f32 stream. A-matrix 16x32 f16 layout (ISA 7.12.2): element j of the
// lane's v16h is channel (j<8 ? j : j+8) + (lane<16 ? 0 : 8).
// ---------------------------------------------------------------------------
__global__ __launch_bounds__(256) void k1_softmax_shuffle(
    const float* __restrict__ kl, float* __restrict__ kh) {
  const int tid  = blockIdx.x * 256 + threadIdx.x;
  const int wid  = tid >> 5;
  const int lane = threadIdx.x & 31;
  const int b    = wid / (NPIX / 16);
  const int p    = (wid % (NPIX / 16)) * 16 + (lane & 15);
  const bool lo  = (lane < 16);

  const float* base = kl + (size_t)b * C_LOG * NPIX + p;

  float v[32];
  float mx = -INFINITY;
#pragma unroll
  for (int a = 0; a < 2; ++a) {
#pragma unroll
    for (int j = 0; j < 16; ++j) {
      int ch = (j < 8 ? j : j + 8) + (lo ? 0 : 8) + a * 32;
      float x = base[(size_t)ch * NPIX];
      v[a * 16 + j] = x;
      mx = fmaxf(mx, x);
    }
  }
  float x64 = base[(size_t)64 * NPIX];
  mx = fmaxf(mx, x64);
  mx = fmaxf(mx, __shfl_xor(mx, 16, 32));

  float e[32];
  v16h a0, a1, ones;
#pragma unroll
  for (int j = 0; j < 16; ++j) {
    e[j]      = __expf(v[j]      - mx);
    e[16 + j] = __expf(v[16 + j] - mx);
    a0[j]   = (_Float16)e[j];
    a1[j]   = (_Float16)e[16 + j];
    ones[j] = (_Float16)1.0f;
  }

  v8f c = {};
  c = __builtin_amdgcn_wmma_f32_16x16x32_f16(false, a0, false, ones, (short)0, c, false, false);
  c = __builtin_amdgcn_wmma_f32_16x16x32_f16(false, a1, false, ones, (short)0, c, false, false);

  const int m = lane & 15;
  float rowsum = extract_row(c, lane, m);
  float inv    = 1.0f / (rowsum + __expf(x64 - mx));

  const int y8 = p / W8_, x8 = p % W8_;
  float* outb = kh + (size_t)b * NPIXOUT;
#pragma unroll
  for (int a = 0; a < 2; ++a) {
#pragma unroll
    for (int j = 0; j < 16; ++j) {
      int ch  = (j < 8 ? j : j + 8) + (lo ? 0 : 8) + a * 32;
      int row = y8 * 8 + (ch >> 3);
      int col = x8 * 8 + (ch & 7);
      outb[row * WOUT + col] = e[a * 16 + j] * inv;
    }
  }
}

// ---------------------------------------------------------------------------
// K2: 5x5 NMS (reduce_window-with--inf-padding semantics) + per-image
// candidate compaction via atomics.
// ---------------------------------------------------------------------------
__global__ __launch_bounds__(256) void k2_nms(
    const float* __restrict__ kh, int* __restrict__ cand, int* __restrict__ cnts) {
  int t = blockIdx.x * 256 + threadIdx.x;
  if (t >= B_ * NPIXOUT) return;
  int b   = t / NPIXOUT;
  int pix = t - b * NPIXOUT;
  int y = pix / WOUT, x = pix % WOUT;
  const float* img = kh + (size_t)b * NPIXOUT;
  float val = img[pix];
  if (!(val > 0.0f)) return;
  int y0 = max(y - 2, 0), y1 = min(y + 2, HOUT - 1);
  int x0 = max(x - 2, 0), x1 = min(x + 2, WOUT - 1);
  float mxv = -INFINITY;
  for (int yy = y0; yy <= y1; ++yy) {
    const float* rowp = img + yy * WOUT;
    __builtin_prefetch(rowp + x0, 0, 0);    // global_prefetch_b8
    for (int xx = x0; xx <= x1; ++xx) mxv = fmaxf(mxv, rowp[xx]);
  }
  if (val == mxv) {
    int slot = atomicAdd(&cnts[b], 1);
    if (slot < NMS_K_) cand[b * NMS_K_ + slot] = pix;
  }
}

// ---------------------------------------------------------------------------
// K3: 5x5 soft-argmax refinement + score. One wave = 16 candidates.
// The three 25-wide reductions (sum, E[x-off], E[y-off]) per candidate are
// computed with three V_WMMA_F32_16X16X32_F16 ops against broadcast-column B
// matrices (ones / offx / offy); window slots 25..31 carry exp=0 in A.
// A element j -> window index k = (j<8 ? j : j+8) + (lane<16 ? 0 : 8).
// B element j -> window index k = (j<8 ? j : j+8) + (lane<16 ? 0 : 8).
// ---------------------------------------------------------------------------
__global__ __launch_bounds__(256) void k3_refine(
    const float* __restrict__ kh, const float* __restrict__ hmap,
    const int* __restrict__ cand, const int* __restrict__ cnts,
    float* __restrict__ refx, float* __restrict__ refy, float* __restrict__ rsc) {
  const int tid  = blockIdx.x * 256 + threadIdx.x;
  const int wid  = tid >> 5;
  const int lane = threadIdx.x & 31;
  const bool lo  = (lane < 16);
  const int b    = wid / (NMS_K_ / 16);
  const int m    = lane & 15;
  const int s    = (wid % (NMS_K_ / 16)) * 16 + m;
  const int t    = b * NMS_K_ + s;

  const int n   = min(cnts[b], NMS_K_);
  const int pix = (s < n) ? cand[t] : 0;     // invalid slots -> corner (pad path)
  const int x = pix % WOUT, y = pix / WOUT;
  const float* img = kh + (size_t)b * NPIXOUT;
  const int x0 = max(x - 2, 0), y0 = max(y - 2, 0);
  const int x1e = min(x + 3, WOUT), y1e = min(y + 3, HOUT);

  // Gather this lane's half of the 5x5 window (values masked to 0 outside the
  // valid sub-window, exactly as the reference), scaled by 1/tau.
  float p[16];
  float pm = 0.0f;                            // heatmap >= 0, masked entries are 0
#pragma unroll
  for (int j = 0; j < 16; ++j) {
    int k = (j < 8 ? j : j + 8) + (lo ? 0 : 8);
    float pk = 0.0f;
    if (k < 25) {
      int i5 = k / 5, j5 = k % 5;
      int rr = min(y0 + i5, HOUT - 1), cc = min(x0 + j5, WOUT - 1);
      bool valid = (y0 + i5 < y1e) && (x0 + j5 < x1e);
      float vv = valid ? img[rr * WOUT + cc] : 0.0f;
      pk = vv * (1.0f / TAU_);
    }
    p[j] = pk;
    pm = fmaxf(pm, pk);
  }
  pm = fmaxf(pm, __shfl_xor(pm, 16, 32));

  v16h a, b1, bx, by;
#pragma unroll
  for (int j = 0; j < 16; ++j) {
    int k  = (j < 8 ? j : j + 8) + (lo ? 0 : 8);
    float ek = (k < 25) ? __expf(p[j] - pm) : 0.0f;
    a[j] = (_Float16)ek;
    int kb = (j < 8 ? j : j + 8) + (lo ? 0 : 8);
    b1[j] = (_Float16)1.0f;
    bx[j] = (_Float16)((kb < 25) ? (float)(kb % 5 - 2) : 0.0f);
    by[j] = (_Float16)((kb < 25) ? (float)(kb / 5 - 2) : 0.0f);
  }

  v8f cs = {}, cx = {}, cy = {};
  cs = __builtin_amdgcn_wmma_f32_16x16x32_f16(false, a, false, b1, (short)0, cs, false, false);
  cx = __builtin_amdgcn_wmma_f32_16x16x32_f16(false, a, false, bx, (short)0, cx, false, false);
  cy = __builtin_amdgcn_wmma_f32_16x16x32_f16(false, a, false, by, (short)0, cy, false, false);

  float sum = extract_row(cs, lane, m);
  float ex  = extract_row(cx, lane, m);
  float ey  = extract_row(cy, lane, m);

  float fx = (float)x0 + ex / sum;
  float fy = (float)y0 + ey / sum;
  if (x == 0 && y == 0) { fx = 0.0f; fy = 0.0f; }   // is_pad path

  if (lo) {
    // nearest on kh (Wi=Hi=800, w=h=800):  ix = kx*800/799 - 0.5
    float ixn = fx * (800.0f / 799.0f) - 0.5f;
    float iyn = fy * (800.0f / 799.0f) - 0.5f;
    int rx = (int)rintf(ixn), ry = (int)rintf(iyn);
    float s_near = (rx >= 0 && rx < WOUT && ry >= 0 && ry < HOUT) ? img[ry * WOUT + rx] : 0.0f;

    // bilinear on heatmap (100x100):  ix = kx*100/799 - 0.5
    float ixb = fx * (100.0f / 799.0f) - 0.5f;
    float iyb = fy * (100.0f / 799.0f) - 0.5f;
    float x0f = floorf(ixb), y0f = floorf(iyb);
    float tx = ixb - x0f, ty = iyb - y0f;
    int bxi = (int)x0f, byi = (int)y0f;
    const float* hb = hmap + (size_t)b * NPIX;
    auto g = [&](int yy, int xx) -> float {
      return (yy >= 0 && yy < H8_ && xx >= 0 && xx < W8_) ? hb[yy * W8_ + xx] : 0.0f;
    };
    float s_bil = g(byi, bxi) * (1 - tx) * (1 - ty) + g(byi, bxi + 1) * tx * (1 - ty) +
                  g(byi + 1, bxi) * (1 - tx) * ty + g(byi + 1, bxi + 1) * tx * ty;

    float sc = s_near * s_bil;
    if ((fx == 0.0f && fy == 0.0f) || s >= n) sc = -1.0f;
    refx[t] = fx; refy[t] = fy; rsc[t] = sc;
  }
}

// ---------------------------------------------------------------------------
// K4: per-image descending bitonic sort of 32768 candidate scores inside
// 256 KB of LDS (CDNA5: 320 KB per workgroup). The load phase is a single
// Tensor Data Mover descriptor: 32768 contiguous f32 -> LDS offset 0,
// issued by wave 0, completion via s_wait_tensorcnt.
// ---------------------------------------------------------------------------
__global__ __launch_bounds__(1024) void k4_sort(
    const float* __restrict__ rsc, const float* __restrict__ refx,
    const float* __restrict__ refy, float* __restrict__ out_scores,
    float* __restrict__ out_kpts) {
  extern __shared__ unsigned char smem[];
  float* sk = (float*)smem;            // 32768 keys at LDS offset 0
  int*   sv = (int*)(sk + NMS_K_);     // 32768 payloads
  const int b = blockIdx.x, tid = threadIdx.x;

#if __has_builtin(__builtin_amdgcn_tensor_load_to_lds)
  if (tid < 32) {
    unsigned long long ga = (unsigned long long)(const void*)(rsc + (size_t)b * NMS_K_);
    // D# group0: count=1, lds_addr=0, global_addr, type=2 (bits 127:126).
    tdm_v4u g0 = { 1u, 0u, (unsigned)(ga & 0xffffffffu),
                   (unsigned)((ga >> 32) & 0x01ffffffu) | (2u << 30) };
    // D# group1: data_size=4B (bits17:16=2); tensor_dim0=32768 (bits 79:48);
    // tensor_dim1=1 (bits 111:80); tile_dim0=32768 (bits 127:112);
    // tile_dim1=1 (bits 143:128); tensor_dim0_stride=32768 (bits 207:160).
    tdm_v8i g1 = { 0x20000, (int)0x80000000u, 0x00010000, (int)0x80000000u,
                   1, 0x8000, 0, 0 };
    tdm_v4i gz = { 0, 0, 0, 0 };
#if __clang_major__ >= 23
    tdm_v8i gz8 = { 0, 0, 0, 0, 0, 0, 0, 0 };
    __builtin_amdgcn_tensor_load_to_lds(g0, g1, gz, gz, gz8, 0);
#else
    __builtin_amdgcn_tensor_load_to_lds(g0, g1, gz, gz, 0);
#endif
    __builtin_amdgcn_s_wait_tensorcnt(0);
  }
  for (int i = tid; i < NMS_K_; i += 1024) sv[i] = i;
#else
  for (int i = tid; i < NMS_K_; i += 1024) { sk[i] = rsc[b * NMS_K_ + i]; sv[i] = i; }
#endif
  __syncthreads();

  for (unsigned k = 2; k <= (unsigned)NMS_K_; k <<= 1)
    for (unsigned j = k >> 1; j > 0; j >>= 1) {
      __syncthreads();
      for (unsigned i = tid; i < (unsigned)NMS_K_; i += 1024) {
        unsigned ixj = i ^ j;
        if (ixj > i) {
          bool desc = ((i & k) == 0);
          float a = sk[i], c2 = sk[ixj];
          bool sw = desc ? (a < c2) : (a > c2);
          if (sw) {
            sk[i] = c2; sk[ixj] = a;
            int tt = sv[i]; sv[i] = sv[ixj]; sv[ixj] = tt;
          }
        }
      }
    }
  __syncthreads();

  for (int r = tid; r < TOP_K_; r += 1024) {
    int slot = sv[r];
    out_scores[b * TOP_K_ + r]         = sk[r];
    out_kpts[(b * TOP_K_ + r) * 2 + 0] = refx[b * NMS_K_ + slot];
    out_kpts[(b * TOP_K_ + r) * 2 + 1] = refy[b * NMS_K_ + slot];
  }
}

// ---------------------------------------------------------------------------
// K5: bicubic (a=-0.75) sampling of 64-channel feats at each kept keypoint
// + L2 normalization. One 64-thread block per keypoint, one thread/channel.
// ---------------------------------------------------------------------------
__device__ inline void cubic_w(float t, float w[4]) {
  const float A = CUBIC_A_;
  float t1 = t + 1.0f;
  w[0] = ((A * t1 - 5.0f * A) * t1 + 8.0f * A) * t1 - 4.0f * A;
  w[1] = ((A + 2.0f) * t - (A + 3.0f)) * t * t + 1.0f;
  float u = 1.0f - t;
  w[2] = ((A + 2.0f) * u - (A + 3.0f)) * u * u + 1.0f;
  w[3] = 1.0f - w[0] - w[1] - w[2];
}

__global__ __launch_bounds__(64) void k5_desc(
    const float* __restrict__ feats, const float* __restrict__ kpts,
    float* __restrict__ fout) {
  __shared__ float red[64];
  const int bk = blockIdx.x;           // b*TOP_K_ + k
  const int c  = threadIdx.x;
  const int b  = bk / TOP_K_;

  float kx = kpts[bk * 2 + 0], ky = kpts[bk * 2 + 1];
  float ix = kx * (100.0f / 799.0f) - 0.5f;
  float iy = ky * (100.0f / 799.0f) - 0.5f;
  float x1f = floorf(ix), y1f = floorf(iy);
  float wx[4], wy[4];
  cubic_w(ix - x1f, wx);
  cubic_w(iy - y1f, wy);
  int x1 = (int)x1f, y1 = (int)y1f;

  const float* fb = feats + ((size_t)b * FEAT_C + c) * NPIX;
  float acc = 0.0f;
#pragma unroll
  for (int i = 0; i < 4; ++i) {
    int yy = y1 + (i - 1);
    float row = 0.0f;
#pragma unroll
    for (int j = 0; j < 4; ++j) {
      int xx = x1 + (j - 1);
      float vv = (yy >= 0 && yy < H8_ && xx >= 0 && xx < W8_) ? fb[yy * W8_ + xx] : 0.0f;
      row += vv * wx[j];
    }
    acc += row * wy[i];
  }

  red[c] = acc * acc;
  __syncthreads();
  for (int s = 32; s > 0; s >>= 1) {
    if (c < s) red[c] += red[c + s];
    __syncthreads();
  }
  float nrm = sqrtf(red[0]);
  fout[(size_t)bk * FEAT_C + c] = acc / fmaxf(nrm, 1e-12f);
}

// ---------------------------------------------------------------------------
extern "C" void kernel_launch(void* const* d_in, const int* in_sizes, int n_in,
                              void* d_out, int out_size, void* d_ws, size_t ws_size,
                              hipStream_t stream) {
  (void)in_sizes; (void)n_in; (void)out_size; (void)ws_size;
  const float* feats = (const float*)d_in[0];   // (32,64,100,100)
  const float* klog  = (const float*)d_in[1];   // (32,65,100,100)
  const float* hmap  = (const float*)d_in[2];   // (32,1,100,100)
  // d_in[3], d_in[4]: h=800, w=800 (compile-time constants here)

  char* ws = (char*)d_ws;
  size_t off = 0;
  auto walloc = [&](size_t bytes) -> char* {
    char* p = ws + off;
    off = (off + bytes + 255) & ~(size_t)255;
    return p;
  };
  float* kh   = (float*)walloc((size_t)B_ * NPIXOUT * sizeof(float)); // 81.9 MB
  int*   cand = (int*)  walloc((size_t)B_ * NMS_K_ * sizeof(int));
  int*   cnts = (int*)  walloc((size_t)B_ * sizeof(int));
  float* refx = (float*)walloc((size_t)B_ * NMS_K_ * sizeof(float));
  float* refy = (float*)walloc((size_t)B_ * NMS_K_ * sizeof(float));
  float* rsc  = (float*)walloc((size_t)B_ * NMS_K_ * sizeof(float));

  float* out_f = (float*)d_out;                          // 32*4096*64
  float* out_s = out_f + (size_t)B_ * TOP_K_ * FEAT_C;   // 32*4096
  float* out_k = out_s + (size_t)B_ * TOP_K_;            // 32*4096*2

  hipMemsetAsync(cnts, 0, B_ * sizeof(int), stream);

  // K1: 20000 waves (16 px each), 8 waves per 256-thread block.
  k1_softmax_shuffle<<<(B_ * (NPIX / 16)) / 8, 256, 0, stream>>>(klog, kh);
  // K2: one thread per kh pixel.
  k2_nms<<<(B_ * NPIXOUT) / 256, 256, 0, stream>>>(kh, cand, cnts);
  // K3: one wave per 16 candidate slots.
  k3_refine<<<(B_ * (NMS_K_ / 16)) / 8, 256, 0, stream>>>(kh, hmap, cand, cnts, refx, refy, rsc);
  // K4: one block per image, 256 KB dynamic LDS, TDM-fed.
  hipFuncSetAttribute(reinterpret_cast<const void*>(k4_sort),
                      hipFuncAttributeMaxDynamicSharedMemorySize, NMS_K_ * 8);
  k4_sort<<<B_, 1024, NMS_K_ * 8, stream>>>(rsc, refx, refy, out_s, out_k);
  // K5: one 64-thread block per kept keypoint.
  k5_desc<<<B_ * TOP_K_, 64, 0, stream>>>(feats, out_k, out_f);
}
